// NeuralMnist_v1_2044404433016
// MI455X (gfx1250) — compile-verified, matrix operand order checked by reading, not run on hardware
//
#include <hip/hip_runtime.h>
#include <hip/hip_bf16.h>
#include <math.h>

// ---------------------------------------------------------------------------
// Fused 3-layer MLP (MNIST) with act_on interventions, WMMA f16->f32 path.
//   B = 65536 rows.  Dims: 784 -> 350 -> 50 -> 10, log_softmax output.
// K-chunk tiled operands (chunks of 32 halfs): every fragment load is a
// 128-bit load with an immediate offset.  Depth-2 A-fragment ring; B is
// ping-pong double-buffered across a manual kt x2 loop body so there is no
// loop-carried B copy.  kt loops kept rolled (#pragma unroll 1) so the
// 64-VGPR accumulator array stays pinned.
// Block = 128 threads (4 waves) covering 128 batch rows (8 M-tiles/wave).
// ---------------------------------------------------------------------------

typedef __attribute__((ext_vector_type(16))) _Float16     v16h;
typedef __attribute__((ext_vector_type(8)))  float        v8f;
typedef __attribute__((ext_vector_type(4)))  unsigned int v4u;

#define ALPHA 0.5f
#define CH    32            // K elements per chunk (one 16x16x32 K-step)
#define ROWS  128           // batch rows per block (8 M-tiles)
#define MT    8             // M-tiles per wave
#define KT1   25            // K-chunks, layer 1 (784 -> 800)
#define KT1P  (KT1 + 1)     // +1 pad chunk for branch-free pipeline
#define KT2   11            // K-chunks, layer 2 (350 -> 352)
#define KT2P  (KT2 + 1)
#define N1T   22            // N-tiles layer 1 (352 cols)
#define D1P   352
#define D2P   64

union F16Frag { v16h v; v4u u[2]; };

// 16-bit A layout (ISA 7.12.2): lane L -> row (L&15); half = L>>4;
// per lane a fragment is two contiguous 16-byte runs.  B is symmetric
// (lane -> column N, halves cover disjoint K ranges).
__device__ __forceinline__ v16h load_frag_p(const _Float16* p) {
  F16Frag f;
  f.u[0] = *(const v4u*)(p);
  f.u[1] = *(const v4u*)(p + 16);
  return f.v;
}

__device__ __forceinline__ v8f wmma_f16(v16h a, v16h b, v8f c) {
  return __builtin_amdgcn_wmma_f32_16x16x32_f16(false, a, false, b,
                                                (short)0, c, false, false);
}

// One K-step: 8 WMMAs against B-fragment `b`, A-ring rotated two ahead.
// aK = current chunk base (lane-adjusted), aKn = next chunk base.
__device__ __forceinline__ void mstep(const _Float16* aK, const _Float16* aKn,
                                      v16h b, v8f acc[MT], v16h& a0, v16h& a1) {
#pragma unroll
  for (int m = 0; m < MT; ++m) {
    v16h anext = (m < MT - 2)
                     ? load_frag_p(aK + (m + 2) * (16 * CH))
                     : load_frag_p(aKn + (m - (MT - 2)) * (16 * CH));
    acc[m] = wmma_f16(a0, b, acc[m]);
    a0 = a1;
    a1 = anext;
  }
}

// ---------------------------------------------------------------------------
// Prep 1: weights -> K-chunk tiled f16, zero padded (incl. pad chunk):
//   w1t[kt][352][32] (kt<26), w2t[kt][64][32] (kt<12)
// ---------------------------------------------------------------------------
__global__ void prep_weights(const float* __restrict__ W1,
                             const float* __restrict__ W2,
                             _Float16* __restrict__ w1t,
                             _Float16* __restrict__ w2t) {
  int i = blockIdx.x * 256 + threadIdx.x;
  const int N1 = KT1P * D1P * CH;           // 292864
  const int N2 = KT2P * D2P * CH;           //  24576
  if (i < N1) {
    int kt = i / (D1P * CH);
    int rem = i % (D1P * CH);
    int n = rem / CH, kk = rem % CH;
    int k = kt * CH + kk;
    float v = (n < 350 && k < 784) ? W1[n * 784 + k] : 0.0f;
    w1t[i] = (_Float16)v;
  } else if (i < N1 + N2) {
    int j = i - N1;
    int kt = j / (D2P * CH);
    int rem = j % (D2P * CH);
    int n = rem / CH, kk = rem % CH;
    int k = kt * CH + kk;
    float v = (n < 50 && k < 350) ? W2[n * 350 + k] : 0.0f;
    w2t[j] = (_Float16)v;
  }
}

// ---------------------------------------------------------------------------
// Prep 2: fused per-column epilogue tables (padded, zero outside range):
//   e0[c] = {ben0, flag}   e1[n] = {b1, ben1, flag, 0}   e2[n] = {b2, ben2, flag, 0}
// ---------------------------------------------------------------------------
__global__ void prep_epi(const float* __restrict__ b1, const float* __restrict__ b2,
                         const float* __restrict__ ben0, const float* __restrict__ ben1,
                         const float* __restrict__ ben2,
                         const int* __restrict__ idx0, const int* __restrict__ idx1,
                         const int* __restrict__ idx2,
                         float2* __restrict__ e0, float4* __restrict__ e1,
                         float4* __restrict__ e2) {
  int t = threadIdx.x;  // 1024 threads, single block
  for (int i = t; i < 800; i += 1024)
    e0[i] = make_float2(i < 784 ? ben0[i] : 0.0f, 0.0f);
  for (int i = t; i < D1P; i += 1024)
    e1[i] = make_float4(i < 350 ? b1[i] : 0.0f, i < 350 ? ben1[i] : 0.0f, 0.0f, 0.0f);
  for (int i = t; i < D2P; i += 1024)
    e2[i] = make_float4(i < 50 ? b2[i] : 0.0f, i < 50 ? ben2[i] : 0.0f, 0.0f, 0.0f);
  __syncthreads();
  if (t < 16)      e0[idx0[t]].y      = 1.0f;
  else if (t < 32) e1[idx1[t - 16]].z = 1.0f;
  else if (t < 40) e2[idx2[t - 32]].z = 1.0f;
}

// ---------------------------------------------------------------------------
// Fused main kernel.  Dynamic LDS:
//   xs  : 26*128*32 halfs = 212992 B   (h2s[128][64] aliases this after L1)
//   h1t : 12*128*32 halfs =  98304 B   -> total 311296 B (<= 320KB WG limit)
// ---------------------------------------------------------------------------
#define LDS_XS (KT1P * ROWS * CH)
#define LDS_H1 (KT2P * ROWS * CH)

__global__ __launch_bounds__(128) void mlp_fused(
    const float* __restrict__ x,
    const _Float16* __restrict__ w1t, const _Float16* __restrict__ w2t,
    const float* __restrict__ W3,     const float* __restrict__ b3,
    const float2* __restrict__ e0, const float4* __restrict__ e1,
    const float4* __restrict__ e2,
    float* __restrict__ out) {
  extern __shared__ _Float16 smem[];
  _Float16* xs  = smem;                    // tiled [26][128][32]
  _Float16* h1t = xs + LDS_XS;             // tiled [12][128][32]
  _Float16* h2s = smem;                    // [128][64], reuses xs after L1

  const int tid  = threadIdx.x;
  const int lane = tid & 31;
  const int wave = tid >> 5;
  const int half = lane >> 4;              // K-half selector for fragments
  const int q    = lane & 15;              // row (A) / column (B) within tile
  const int rowBase = blockIdx.x * ROWS;

  // ---- Stage x: act0 fused, f32 -> f16, tiled, zero-pad cols 784..831 ----
  for (int c0 = 0; c0 < 832; c0 += 128) {
    const int c = c0 + tid;
    float ben = 0.0f, flg = 0.0f;
    if (c < 784) { float2 e = e0[c]; ben = e.x; flg = e.y; }
    const int kt = c >> 5, kk = c & 31;
    for (int r = 0; r < ROWS; ++r) {
      float v = 0.0f;
      if (c < 784) {
        v = __builtin_nontemporal_load(&x[(size_t)(rowBase + r) * 784 + c]);
        float d = v - ben;
        v = (flg != 0.0f && fabsf(d) < 1.0f) ? v - ALPHA * d : v;
      }
      if (c < 832) xs[(kt * ROWS + r) * CH + kk] = (_Float16)v;
    }
  }
  __syncthreads();

  // ---- Layer 1: h1[128x352] = relu(xs @ W1^T); waves split 22 N-tiles ----
  const _Float16* aBase1 = xs + q * CH + half * 8;
  for (int nt = wave; nt < N1T; nt += 4) {
    const int nCol = nt * 16 + q;
    const _Float16* bBase = w1t + (size_t)nCol * CH + half * 8;
    v8f acc[MT] = {};
    const _Float16* aP = aBase1;                     // current kt chunk base
    v16h bcur = load_frag_p(bBase);                  // B(0)
    v16h a0   = load_frag_p(aP);                     // A(0,0)
    v16h a1   = load_frag_p(aP + 16 * CH);           // A(0,1)
#pragma unroll 1
    for (int kt = 0; kt < KT1 - 1; kt += 2) {
      const _Float16* bK1 = bBase + (size_t)(kt + 1) * (D1P * CH);
      __builtin_prefetch(bK1 + D1P * CH, 0, 1);
      v16h bnext = load_frag_p(bK1);                 // B(kt+1), ping
      mstep(aP, aP + ROWS * CH, bcur, acc, a0, a1);  // K-step kt
      // bcur is dead here: B(kt+2) can land in its registers (no copy).
      v16h bnext2 = load_frag_p(bK1 + D1P * CH);     // B(kt+2), pong
      mstep(aP + ROWS * CH, aP + 2 * (ROWS * CH), bnext, acc, a0, a1); // kt+1
      bcur = bnext2;
      aP += 2 * (ROWS * CH);
    }
    mstep(aP, aP + ROWS * CH, bcur, acc, a0, a1);    // tail K-step (kt = 24)
    const float4 e = e1[nCol];
    _Float16* hB = h1t + ((nCol >> 5) * ROWS) * CH + (nCol & 31);
#pragma unroll
    for (int m = 0; m < MT; ++m) {
#pragma unroll
      for (int r = 0; r < 8; ++r) {
        const int M = m * 16 + r + half * 8;
        float v = fmaxf(acc[m][r] + e.x, 0.0f);
        float d = v - e.y;
        v = (e.z != 0.0f && fabsf(d) < 1.0f) ? v - ALPHA * d : v;
        hB[M * CH] = (_Float16)v;
      }
    }
  }
  __syncthreads();

  // ---- Layer 2: h2[128x64] = relu(h1t @ W2^T); wave w owns N-tile w ------
  {
    const int nCol = wave * 16 + q;
    const _Float16* aBase2 = h1t + q * CH + half * 8;
    const _Float16* bBase = w2t + (size_t)nCol * CH + half * 8;
    v8f acc[MT] = {};
    const _Float16* aP = aBase2;
    v16h bcur = load_frag_p(bBase);
    v16h a0   = load_frag_p(aP);
    v16h a1   = load_frag_p(aP + 16 * CH);
#pragma unroll 1
    for (int kt = 0; kt < KT2 - 1; kt += 2) {
      const _Float16* bK1 = bBase + (size_t)(kt + 1) * (D2P * CH);
      v16h bnext = load_frag_p(bK1);                 // B(kt+1)
      mstep(aP, aP + ROWS * CH, bcur, acc, a0, a1);  // kt
      v16h bnext2 = load_frag_p(bK1 + D2P * CH);     // B(kt+2)
      mstep(aP + ROWS * CH, aP + 2 * (ROWS * CH), bnext, acc, a0, a1); // kt+1
      bcur = bnext2;
      aP += 2 * (ROWS * CH);
    }
    mstep(aP, aP + ROWS * CH, bcur, acc, a0, a1);    // tail K-step (kt = 10)
    const float4 e = e2[nCol];
    // L1's xs region is dead now; results land in h2s (aliases xs).
#pragma unroll
    for (int m = 0; m < MT; ++m) {
#pragma unroll
      for (int r = 0; r < 8; ++r) {
        const int M = m * 16 + r + half * 8;
        float v = fmaxf(acc[m][r] + e.x, 0.0f);
        float d = v - e.y;
        v = (e.z != 0.0f && fabsf(d) < 1.0f) ? v - ALPHA * d : v;
        h2s[M * D2P + nCol] = (_Float16)v;
      }
    }
  }
  __syncthreads();

  // ---- Layer 3 (128x50 @ 50x10) + log_softmax: thread t owns row t -------
  {
    const int M = tid;
    float h[50];
#pragma unroll
    for (int k = 0; k < 50; ++k) h[k] = (float)h2s[M * D2P + k];
    float logits[10];
#pragma unroll
    for (int j = 0; j < 10; ++j) {
      float s = b3[j];
      for (int k = 0; k < 50; ++k) s += h[k] * W3[j * 50 + k];
      logits[j] = s;
    }
    float mx = logits[0];
#pragma unroll
    for (int j = 1; j < 10; ++j) mx = fmaxf(mx, logits[j]);
    float se = 0.0f;
#pragma unroll
    for (int j = 0; j < 10; ++j) se += expf(logits[j] - mx);
    const float lse = logf(se) + mx;
    float* o = out + (size_t)(rowBase + M) * 10;
#pragma unroll
    for (int j = 0; j < 10; ++j)
      __builtin_nontemporal_store(logits[j] - lse, &o[j]);
  }
}

// ---------------------------------------------------------------------------
// Launch. Inputs (setup_inputs order):
//  0:x 1:W1 2:b1 3:W2 4:b2 5:W3 6:b3 7:ben0 8:ben1 9:ben2 10:idx0 11:idx1 12:idx2
// ---------------------------------------------------------------------------
extern "C" void kernel_launch(void* const* d_in, const int* in_sizes, int n_in,
                              void* d_out, int out_size, void* d_ws, size_t ws_size,
                              hipStream_t stream) {
  const float* x    = (const float*)d_in[0];
  const float* W1   = (const float*)d_in[1];
  const float* b1   = (const float*)d_in[2];
  const float* W2   = (const float*)d_in[3];
  const float* b2   = (const float*)d_in[4];
  const float* W3   = (const float*)d_in[5];
  const float* b3   = (const float*)d_in[6];
  const float* ben0 = (const float*)d_in[7];
  const float* ben1 = (const float*)d_in[8];
  const float* ben2 = (const float*)d_in[9];
  const int* idx0   = (const int*)d_in[10];
  const int* idx1   = (const int*)d_in[11];
  const int* idx2   = (const int*)d_in[12];

  char* ws = (char*)d_ws;
  const size_t w1tBytes = (size_t)KT1P * D1P * CH * 2;   // 585728
  const size_t w2tBytes = (size_t)KT2P * D2P * CH * 2;   //  49152
  _Float16* w1t = (_Float16*)ws;
  _Float16* w2t = (_Float16*)(ws + w1tBytes);
  float2*   e0  = (float2*)(ws + w1tBytes + w2tBytes);                 // 6400 B
  float4*   e1  = (float4*)(ws + w1tBytes + w2tBytes + 800 * 8);       // 5632 B
  float4*   e2  = (float4*)(ws + w1tBytes + w2tBytes + 800 * 8 + D1P * 16);

  float* out = (float*)d_out;

  const int prepN = KT1P * D1P * CH + KT2P * D2P * CH;   // 317440
  prep_weights<<<(prepN + 255) / 256, 256, 0, stream>>>(W1, W2, w1t, w2t);
  prep_epi<<<1, 1024, 0, stream>>>(b1, b2, ben0, ben1, ben2,
                                   idx0, idx1, idx2, e0, e1, e2);

  const size_t shm = (size_t)(LDS_XS + LDS_H1) * sizeof(_Float16);     // 311296
  mlp_fused<<<65536 / ROWS, 128, shm, stream>>>(
      x, w1t, w2t, W3, b3, e0, e1, e2, out);
}
